// GNet_EMA_61168924230388
// MI455X (gfx1250) — compile-verified, matrix operand order checked by reading, not run on hardware
//
#include <hip/hip_runtime.h>
#include <math.h>

// ---------------- problem constants (from reference) ----------------
#define Bt 4
#define Nn 1536
#define FIN 8
#define LD 128
#define K0 1382    // max(2, int(0.9*1536))
#define K1 967     // max(2, int(0.7*1382))
#define K0p 1408   // K0 padded to 64
#define K1p 1024   // K1 padded to 64

typedef __bf16 bf16_t;
typedef __bf16 v8bf  __attribute__((ext_vector_type(8)));
typedef __bf16 v16bf __attribute__((ext_vector_type(16)));
typedef float  v8f   __attribute__((ext_vector_type(8)));

// =====================================================================
// WMMA GEMM, fully padded dims (M%64==0, N%64==0, K%32==0), no guards.
//   C = ACT(A[MxK] @ B[KxN] + bias) (+ residual)
// LDS tiles stored pre-permuted into the wave32 WMMA fragment layout
// (ISA 7.12.2): each staging thread converts 8 floats with packed
// v_cvt_pk_bf16_f32 and writes one ds_store_b128; each lane reads its
// whole fragment as one 32B vector (2x ds_load_b128). Double-buffered
// LDS -> one barrier per K-chunk, staging of chunk k+1 overlaps the
// WMMAs of chunk k. 8 waves, each owns a 16x32 slab -> 2 wmma/chunk.
// Template params: BIN=binarize operands, ACT 0=none 1=elu
// 3=threshold(!=0), HASB=bias, HASR=residual.
// =====================================================================
template<int BIN, int ACT, int HASB, int HASR>
__global__ __launch_bounds__(256)
void gemm_wmma_kernel(const float* __restrict__ A, int lda, long long sA,
                      const float* __restrict__ Bm, int ldb, long long sB,
                      float* __restrict__ C, int ldc, long long sC,
                      const float* __restrict__ bias,
                      const float* __restrict__ Res, long long sR,
                      int K)
{
    __shared__ __align__(32) bf16_t Asf[2][4][32][16];   // double buffered
    __shared__ __align__(32) bf16_t Bsf[2][4][32][16];

    const int b = blockIdx.z;
    A  += (long long)b * sA;
    Bm += (long long)b * sB;
    C  += (long long)b * sC;
    if (HASR) Res += (long long)b * sR;

    const int tid  = threadIdx.x;
    const int wave = tid >> 5;
    const int lane = tid & 31;
    const int wm   = wave & 3;                 // wave row tile (0..3)
    const int wn   = wave >> 2;                // wave col pair (0..1)
    const long long rowBase = (long long)blockIdx.y * 64;
    const long long colBase = (long long)blockIdx.x * 64;

    // staging coords: A thread -> (row, k-octet); B thread -> (col, k-octet)
    const int ar = tid >> 2, ag = tid & 3;
    const int atile = ar >> 4, amrow = ar & 15;
    const int alane = amrow + ((ag & 1) ? 16 : 0);
    const int ae0   = (ag >= 2) ? 8 : 0;
    const float* aptr = A + (rowBase + ar) * lda + ag * 8;

    const int bn = tid & 63, bg = tid >> 6;
    const int btile = bn >> 4, bnl = bn & 15;
    const int blane = bnl + ((bg & 1) ? 16 : 0);
    const int be0   = (bg >= 2) ? 8 : 0;
    const float* bptr = Bm + (long long)(bg * 8) * ldb + colBase + bn;

    auto stage = [&](int buf, int kk) {
        // A: 2x global_load_b128 -> packed cvt -> 1x ds_store_b128
        float4 x0 = *(const float4*)(aptr + kk);
        float4 x1 = *(const float4*)(aptr + kk + 4);
        __builtin_prefetch(aptr + kk + 32, 0, 1);   // global_prefetch_b8
        float xs[8] = {x0.x, x0.y, x0.z, x0.w, x1.x, x1.y, x1.z, x1.w};
        v8bf pa;
#pragma unroll
        for (int j = 0; j < 8; ++j) {
            float x = xs[j];
            if (BIN) x = (x != 0.0f) ? 1.0f : 0.0f;
            pa[j] = (bf16_t)x;
        }
        // B: 8 k-strided b32 loads (coalesced across lanes) -> 1 store
        const float* bp = bptr + (long long)kk * ldb;
        v8bf pb;
#pragma unroll
        for (int j = 0; j < 8; ++j) {
            float x = bp[(long long)j * ldb];
            if (BIN) x = (x != 0.0f) ? 1.0f : 0.0f;
            pb[j] = (bf16_t)x;
        }
        *(v8bf*)&Asf[buf][atile][alane][ae0] = pa;
        *(v8bf*)&Bsf[buf][btile][blane][be0] = pb;
    };

    v8f acc0 = {}, acc1 = {};
    stage(0, 0);
    int cur = 0;
    for (int kk = 0; kk < K; kk += 32) {
        __syncthreads();
        if (kk + 32 < K) stage(cur ^ 1, kk + 32);
        v16bf af  = *(const v16bf*)&Asf[cur][wm][lane][0];
        v16bf bf0 = *(const v16bf*)&Bsf[cur][wn * 2 + 0][lane][0];
        v16bf bf1 = *(const v16bf*)&Bsf[cur][wn * 2 + 1][lane][0];
        acc0 = __builtin_amdgcn_wmma_f32_16x16x32_bf16(false, af, false, bf0,
                                                       (short)0, acc0, false, false);
        acc1 = __builtin_amdgcn_wmma_f32_16x16x32_bf16(false, af, false, bf1,
                                                       (short)0, acc1, false, false);
        cur ^= 1;
    }

    // ---- epilogue (straight-line: everything compile-time) --------
    const int mo = (lane >> 4) * 8;
    const long long m0 = rowBase + wm * 16 + mo;
    const long long n0 = colBase + wn * 32 + (lane & 15);
    const long long n1 = n0 + 16;
    const float b0 = HASB ? bias[n0] : 0.0f;
    const float b1 = HASB ? bias[n1] : 0.0f;
#pragma unroll
    for (int r = 0; r < 8; ++r) {
        long long m = m0 + r;
        float x0 = acc0[r] + b0;
        float x1 = acc1[r] + b1;
        if (ACT == 1) {
            x0 = (x0 > 0.0f) ? x0 : (__expf(x0) - 1.0f);
            x1 = (x1 > 0.0f) ? x1 : (__expf(x1) - 1.0f);
        } else if (ACT == 3) {
            x0 = (x0 > 0.5f) ? 1.0f : 0.0f;
            x1 = (x1 > 0.5f) ? 1.0f : 0.0f;
        }
        if (HASR) {
            x0 += Res[m * ldc + n0];
            x1 += Res[m * ldc + n1];
        }
        C[m * ldc + n0] = x0;
        C[m * ldc + n1] = x1;
    }
}

// ---------------- row sums (for column normalization) ---------------
__global__ __launch_bounds__(256)
void rowsum_kernel(const float* __restrict__ A, int n, float* __restrict__ rs)
{
    const int b = blockIdx.y, row = blockIdx.x;
    if (row >= n) return;
    A += (long long)b * Nn * Nn + (long long)row * Nn;
    __shared__ float red[256];
    float s = 0.0f;
    for (int j = threadIdx.x; j < n; j += 256) s += A[j];
    red[threadIdx.x] = s;
    __syncthreads();
    for (int o = 128; o > 0; o >>= 1) {
        if (threadIdx.x < o) red[threadIdx.x] += red[threadIdx.x + o];
        __syncthreads();
    }
    if (threadIdx.x == 0) rs[(long long)b * Nn + row] = red[0];
}

// G[i][j] = src[i][j] / rs[j]   (in-place safe)
__global__ void colscale_kernel(float* __restrict__ G,
                                const float* __restrict__ src,
                                int n, const float* __restrict__ rs)
{
    const int b = blockIdx.z;
    const int j = blockIdx.x * 16 + threadIdx.x;
    const int i = blockIdx.y * 16 + threadIdx.y;
    if (i < n && j < n) {
        long long o = (long long)b * Nn * Nn + (long long)i * Nn + j;
        G[o] = src[o] / rs[(long long)b * Nn + j];
    }
}

// zero an npad x npad region (npad % 16 == 0)
__global__ void zero_mat_kernel(float* __restrict__ X)
{
    const int b = blockIdx.z;
    const long long i = blockIdx.y * 16 + threadIdx.y;
    const long long j = blockIdx.x * 16 + threadIdx.x;
    X[(long long)b * Nn * Nn + i * Nn + j] = 0.0f;
}

// zero feature rows [start, start+gridDim.x)
__global__ void zero_rows_kernel(float* __restrict__ X, int start)
{
    const int b = blockIdx.y;
    const long long i = start + blockIdx.x;
    X[((long long)b * Nn + i) * LD + threadIdx.x] = 0.0f;
}

// ---------------- zero-padded copies for the F_IN=8 GEMM ------------
__global__ void pad_hs_kernel(const float* __restrict__ hs, float* __restrict__ o)
{
    const int b = blockIdx.y, i = blockIdx.x, c = threadIdx.x; // block 32
    o[((long long)b * Nn + i) * 32 + c] =
        (c < FIN) ? hs[((long long)b * Nn + i) * FIN + c] : 0.0f;
}
__global__ void pad_sw_kernel(const float* __restrict__ sW, float* __restrict__ o)
{
    const int k = blockIdx.x, c = threadIdx.x;                 // 32 x 128
    o[k * LD + c] = (k < FIN) ? sW[k * LD + c] : 0.0f;
}

// ---------------- pooling scores: sigmoid(h @ pW + pb) --------------
__global__ void score_kernel(const float* __restrict__ H,
                             const float* __restrict__ pW,
                             const float* __restrict__ pb,
                             int n, float* __restrict__ sc)
{
    const int b = blockIdx.y;
    const int i = blockIdx.x * 256 + threadIdx.x;
    if (i >= n) return;
    const float* h = H + ((long long)b * Nn + i) * LD;
    float a = pb[0];
#pragma unroll 8
    for (int c = 0; c < LD; ++c) a += h[c] * pW[c];
    sc[(long long)b * Nn + i] = 1.0f / (1.0f + __expf(-a));
}

// ---------------- top-k: per-graph bitonic sort (stable) ------------
__global__ __launch_bounds__(1024)
void topk_kernel(const float* __restrict__ sc, int n, int k,
                 int* __restrict__ idx_out, float* __restrict__ val_out)
{
    __shared__ float v[2048];
    __shared__ int   id[2048];
    const int b = blockIdx.x;
    sc += (long long)b * Nn;
    const int t = threadIdx.x;
    for (int i = t; i < 2048; i += 1024) {
        if (i < n) { v[i] = sc[i]; id[i] = i; }
        else       { v[i] = -3.0e38f; id[i] = 0x7FFFFFFF; }
    }
    for (int size = 2; size <= 2048; size <<= 1) {
        for (int stride = size >> 1; stride > 0; stride >>= 1) {
            __syncthreads();
            int lo = 2 * t - (t & (stride - 1));
            int hi = lo + stride;
            bool desc = ((lo & size) == 0);
            float av = v[lo], bv = v[hi];
            int   ai = id[lo], bi = id[hi];
            bool aFirst = (av > bv) || (av == bv && ai < bi);
            if (desc ? !aFirst : aFirst) {
                v[lo] = bv; v[hi] = av; id[lo] = bi; id[hi] = ai;
            }
        }
    }
    __syncthreads();
    for (int i = t; i < k; i += 1024) {
        idx_out[(long long)b * Nn + i] = id[i];
        val_out[(long long)b * Nn + i] = v[i];
    }
}

// ---------------- gathers / scatter / elementwise -------------------
__global__ void gather_rows_kernel(const float* __restrict__ H,
                                   float* __restrict__ O,
                                   const int* __restrict__ idx,
                                   const float* __restrict__ vals, int k)
{
    const int b = blockIdx.y, i = blockIdx.x;
    if (i >= k) return;
    const int c = threadIdx.x;
    const int r = idx[(long long)b * Nn + i];
    O[((long long)b * Nn + i) * LD + c] =
        H[((long long)b * Nn + r) * LD + c] * vals[(long long)b * Nn + i];
}

__global__ void gather_adj_kernel(const float* __restrict__ U,
                                  float* __restrict__ S,
                                  const int* __restrict__ idx, int k)
{
    const int b = blockIdx.z;
    const int j = blockIdx.x * 16 + threadIdx.x;
    const int i = blockIdx.y * 16 + threadIdx.y;
    if (i < k && j < k) {
        int ri = idx[(long long)b * Nn + i];
        int rj = idx[(long long)b * Nn + j];
        S[(long long)b * Nn * Nn + (long long)i * Nn + j] =
            U[(long long)b * Nn * Nn + (long long)ri * Nn + rj];
    }
}

__global__ void scatter_rows_kernel(const float* __restrict__ H,
                                    float* __restrict__ O,
                                    const int* __restrict__ idx, int k)
{
    const int b = blockIdx.y, i = blockIdx.x;
    if (i >= k) return;
    const int r = idx[(long long)b * Nn + i];
    O[((long long)b * Nn + r) * LD + threadIdx.x] =
        H[((long long)b * Nn + i) * LD + threadIdx.x];
}

__global__ void add_kernel(float* __restrict__ X, const float* __restrict__ Y,
                           long long n)
{
    long long i = (long long)blockIdx.x * 256 + threadIdx.x;
    if (i < n) X[i] += Y[i];
}

// ---------------- heads: out = hid @ W2 + b2 (nout in {3,1}) --------
__global__ void head_kernel(const float* __restrict__ HID,
                            const float* __restrict__ W2,
                            const float* __restrict__ b2,
                            int nout, int off, float* __restrict__ out)
{
    const int b = blockIdx.y;
    const int i = blockIdx.x * 256 + threadIdx.x;
    if (i >= Nn) return;
    const float* h = HID + ((long long)b * Nn + i) * LD;
    for (int o = 0; o < nout; ++o) {
        float a = b2[o];
#pragma unroll 8
        for (int c = 0; c < LD; ++c) a += h[c] * W2[c * nout + o];
        out[1 + ((long long)b * Nn + i) * 4 + off + o] = a;
    }
}

// ---------------- masked MSE loss, single-WG deterministic ----------
__global__ __launch_bounds__(1024)
void loss_kernel(const float* __restrict__ hs, const float* __restrict__ tg,
                 float* __restrict__ out)
{
    __shared__ float r0[1024], r1[1024], r2[1024], r3[1024];
    float sdv = 0.0f, sds = 0.0f, cv = 0.0f, cs = 0.0f;
    for (long long t = threadIdx.x; t < (long long)Bt * Nn; t += 1024) {
        float nt = hs[t * FIN + 3];
        bool mv = (nt == 0.0f);
        bool ms = (nt == 0.0f) || (nt == 6.0f);
        const float* p  = out + 1 + t * 4;
        const float* tt = tg + t * FIN;
        if (mv) {
            for (int o = 0; o < 3; ++o) { float d = p[o] - tt[4 + o]; sdv += d * d; }
            cv += 1.0f;
        }
        if (ms) { float d = p[3] - tt[7]; sds += d * d; cs += 1.0f; }
    }
    r0[threadIdx.x] = sdv; r1[threadIdx.x] = sds;
    r2[threadIdx.x] = cv;  r3[threadIdx.x] = cs;
    __syncthreads();
    for (int o = 512; o > 0; o >>= 1) {
        if (threadIdx.x < o) {
            r0[threadIdx.x] += r0[threadIdx.x + o];
            r1[threadIdx.x] += r1[threadIdx.x + o];
            r2[threadIdx.x] += r2[threadIdx.x + o];
            r3[threadIdx.x] += r3[threadIdx.x + o];
        }
        __syncthreads();
    }
    if (threadIdx.x == 0) {
        float nv = fmaxf(r2[0] * 3.0f, 1.0f);
        float ns = fmaxf(r3[0], 1.0f);
        out[0] = r0[0] / nv + r1[0] / ns;
    }
}

// =====================================================================
extern "C" void kernel_launch(void* const* d_in, const int* in_sizes, int n_in,
                              void* d_out, int out_size, void* d_ws, size_t ws_size,
                              hipStream_t stream)
{
    (void)in_sizes; (void)n_in; (void)out_size; (void)ws_size;

    const float* gs  = (const float*)d_in[0];
    const float* hs  = (const float*)d_in[1];
    const float* tg  = (const float*)d_in[2];
    const float* sW  = (const float*)d_in[3];
    const float* sb  = (const float*)d_in[4];
    const float* dW0 = (const float*)d_in[5],  *db0 = (const float*)d_in[6];
    const float* uW0 = (const float*)d_in[7],  *ub0 = (const float*)d_in[8];
    const float* pW0 = (const float*)d_in[9],  *pb0 = (const float*)d_in[10];
    const float* dW1 = (const float*)d_in[11], *db1 = (const float*)d_in[12];
    const float* uW1 = (const float*)d_in[13], *ub1 = (const float*)d_in[14];
    const float* pW1 = (const float*)d_in[15], *pb1 = (const float*)d_in[16];
    const float* bW  = (const float*)d_in[17], *bb  = (const float*)d_in[18];
    const float* vW1 = (const float*)d_in[19], *vb1 = (const float*)d_in[20];
    const float* vW2 = (const float*)d_in[21], *vb2 = (const float*)d_in[22];
    const float* tW1 = (const float*)d_in[23], *tb1 = (const float*)d_in[24];
    const float* tW2 = (const float*)d_in[25], *tb2 = (const float*)d_in[26];
    float* out = (float*)d_out;

    // ---- workspace layout (f32 elements) ----
    const long long MAT  = (long long)Bt * Nn * Nn;
    const long long FEAT = (long long)Bt * Nn * LD;
    const long long VEC  = (long long)Bt * Nn;
    float* w = (float*)d_ws;
    float* GN0  = w;              // normalized original adjacency
    float* G1   = GN0 + MAT;      // level-1 adjacency (K0p x K0p, stride Nn)
    float* G2   = G1 + MAT;       // level-2 adjacency (K1p x K1p, stride Nn)
    float* U2   = G2 + MAT;       // two-hop scratch
    float* H0   = U2 + MAT;       // org_h
    float* TMP  = H0 + FEAT;      // h @ W temporaries
    float* OUT0 = TMP + FEAT;     // outs[0]
    float* OUT1 = OUT0 + FEAT;    // outs[1]
    float* HP   = OUT1 + FEAT;    // pooled features
    float* HC   = HP + FEAT;      // current features (bottom / unpool)
    float* UNP  = HC + FEAT;      // unpool scatter target
    float* HN   = UNP + FEAT;     // h_nodes
    float* HID  = HN + FEAT;      // head hidden
    float* HSP  = HID + FEAT;     // hs zero-padded to Bt x Nn x 32
    float* SWP  = HSP + (long long)Bt * Nn * 32;   // sW padded to 32 x 128
    float* RS   = SWP + 32 * LD;  // row sums
    float* SC   = RS + VEC;       // scores
    float* VALS = SC + VEC;       // sorted top-k values
    int*   IDX0 = (int*)(VALS + VEC);
    int*   IDX1 = IDX0 + VEC;

    const long long sM = (long long)Nn * Nn;
    const long long sF = (long long)Nn * LD;

    // dispatch to the 4 template instantiations actually used
    auto gemm = [&](const float* A, int lda, long long sA,
                    const float* Bp, int ldb, long long sB,
                    float* C, int ldc, long long sC,
                    const float* bias, const float* Res, long long sR,
                    int M, int N, int K, int binA, int act) {
        dim3 g(N / 64, M / 64, Bt);
        if (binA)
            gemm_wmma_kernel<1, 3, 0, 0><<<g, 256, 0, stream>>>(
                A, lda, sA, Bp, ldb, sB, C, ldc, sC, bias, Res, sR, K);
        else if (Res)
            gemm_wmma_kernel<0, 1, 1, 1><<<g, 256, 0, stream>>>(
                A, lda, sA, Bp, ldb, sB, C, ldc, sC, bias, Res, sR, K);
        else if (bias)
            gemm_wmma_kernel<0, 1, 1, 0><<<g, 256, 0, stream>>>(
                A, lda, sA, Bp, ldb, sB, C, ldc, sC, bias, Res, sR, K);
        else
            gemm_wmma_kernel<0, 0, 0, 0><<<g, 256, 0, stream>>>(
                A, lda, sA, Bp, ldb, sB, C, ldc, sC, bias, Res, sR, K);
    };
    auto normalize = [&](float* dst, const float* src, int n) {
        rowsum_kernel<<<dim3(n, Bt), 256, 0, stream>>>(src, n, RS);
        dim3 g((n + 15) / 16, (n + 15) / 16, Bt);
        colscale_kernel<<<g, dim3(16, 16), 0, stream>>>(dst, src, n, RS);
    };

    // ---- g = norm(a) ----
    normalize(GN0, gs, Nn);

    // ---- h0 = elu(g @ (hs @ sW) + sb) (padded K: 8 -> 32) ----
    pad_hs_kernel<<<dim3(Nn, Bt), 32, 0, stream>>>(hs, HSP);
    pad_sw_kernel<<<32, LD, 0, stream>>>(sW, SWP);
    gemm(HSP, 32, (long long)Nn * 32, SWP, LD, 0, TMP, LD, sF,
         nullptr, nullptr, 0, Nn, LD, 32, 0, 0);
    gemm(GN0, Nn, sM, TMP, LD, sF, H0, LD, sF, sb, nullptr, 0, Nn, LD, Nn, 0, 1);

    // ---- level 0: out0 = elu(g @ (h0 @ dW0) + db0) ----
    gemm(H0, LD, sF, dW0, LD, 0, TMP, LD, sF, nullptr, nullptr, 0, Nn, LD, LD, 0, 0);
    gemm(GN0, Nn, sM, TMP, LD, sF, OUT0, LD, sF, db0, nullptr, 0, Nn, LD, Nn, 0, 1);

    score_kernel<<<dim3((Nn + 255) / 256, Bt), 256, 0, stream>>>(OUT0, pW0, pb0, Nn, SC);
    topk_kernel<<<Bt, 1024, 0, stream>>>(SC, Nn, K0, IDX0, VALS);
    gather_rows_kernel<<<dim3(K0, Bt), LD, 0, stream>>>(OUT0, HP, IDX0, VALS, K0);
    zero_rows_kernel<<<dim3(K0p - K0, Bt), LD, 0, stream>>>(HP, K0);

    // two-hop: U2 = ((g!=0)@(g!=0))!=0 ; G1 = norm(U2[idx0][:,idx0])
    gemm(GN0, Nn, sM, GN0, Nn, sM, U2, Nn, sM, nullptr, nullptr, 0, Nn, Nn, Nn, 1, 3);
    zero_mat_kernel<<<dim3(K0p / 16, K0p / 16, Bt), dim3(16, 16), 0, stream>>>(G1);
    gather_adj_kernel<<<dim3((K0 + 15) / 16, (K0 + 15) / 16, Bt), dim3(16, 16),
                        0, stream>>>(U2, G1, IDX0, K0);
    normalize(G1, G1, K0);

    // ---- level 1: out1 = elu(g1 @ (hp @ dW1) + db1) ----
    gemm(HP, LD, sF, dW1, LD, 0, TMP, LD, sF, nullptr, nullptr, 0, K0p, LD, LD, 0, 0);
    gemm(G1, Nn, sM, TMP, LD, sF, OUT1, LD, sF, db1, nullptr, 0, K0p, LD, K0p, 0, 1);

    score_kernel<<<dim3((K0 + 255) / 256, Bt), 256, 0, stream>>>(OUT1, pW1, pb1, K0, SC);
    topk_kernel<<<Bt, 1024, 0, stream>>>(SC, K0, K1, IDX1, VALS);
    gather_rows_kernel<<<dim3(K1, Bt), LD, 0, stream>>>(OUT1, HP, IDX1, VALS, K1);
    zero_rows_kernel<<<dim3(K1p - K1, Bt), LD, 0, stream>>>(HP, K1);

    gemm(G1, Nn, sM, G1, Nn, sM, U2, Nn, sM, nullptr, nullptr, 0, K0p, K0p, K0p, 1, 3);
    zero_mat_kernel<<<dim3(K1p / 16, K1p / 16, Bt), dim3(16, 16), 0, stream>>>(G2);
    gather_adj_kernel<<<dim3((K1 + 15) / 16, (K1 + 15) / 16, Bt), dim3(16, 16),
                        0, stream>>>(U2, G2, IDX1, K1);
    normalize(G2, G2, K1);

    // ---- bottom: hc = elu(g2 @ (hp @ bW) + bb) ----
    gemm(HP, LD, sF, bW, LD, 0, TMP, LD, sF, nullptr, nullptr, 0, K1p, LD, LD, 0, 0);
    gemm(G2, Nn, sM, TMP, LD, sF, HC, LD, sF, bb, nullptr, 0, K1p, LD, K1p, 0, 1);

    // ---- unpool i=0 (j=1) ----
    zero_rows_kernel<<<dim3(K0p, Bt), LD, 0, stream>>>(UNP, 0);
    scatter_rows_kernel<<<dim3(K1, Bt), LD, 0, stream>>>(HC, UNP, IDX1, K1);
    gemm(UNP, LD, sF, uW0, LD, 0, TMP, LD, sF, nullptr, nullptr, 0, K0p, LD, LD, 0, 0);
    gemm(G1, Nn, sM, TMP, LD, sF, HC, LD, sF, ub0, OUT1, sF, K0p, LD, K0p, 0, 1);

    // ---- unpool i=1 (j=0) ----
    zero_rows_kernel<<<dim3(Nn, Bt), LD, 0, stream>>>(UNP, 0);
    scatter_rows_kernel<<<dim3(K0, Bt), LD, 0, stream>>>(HC, UNP, IDX0, K0);
    gemm(UNP, LD, sF, uW1, LD, 0, TMP, LD, sF, nullptr, nullptr, 0, Nn, LD, LD, 0, 0);
    gemm(GN0, Nn, sM, TMP, LD, sF, HN, LD, sF, ub1, OUT0, sF, Nn, LD, Nn, 0, 1);

    // ---- h_nodes = hn + org_h ----
    add_kernel<<<(int)((FEAT + 255) / 256), 256, 0, stream>>>(HN, H0, FEAT);

    // ---- heads ----
    gemm(HN, LD, sF, vW1, LD, 0, HID, LD, sF, vb1, nullptr, 0, Nn, LD, LD, 0, 1);
    head_kernel<<<dim3((Nn + 255) / 256, Bt), 256, 0, stream>>>(HID, vW2, vb2, 3, 0, out);
    gemm(HN, LD, sF, tW1, LD, 0, HID, LD, sF, tb1, nullptr, 0, Nn, LD, LD, 0, 1);
    head_kernel<<<dim3((Nn + 255) / 256, Bt), 256, 0, stream>>>(HID, tW2, tb2, 1, 3, out);

    // ---- loss ----
    loss_kernel<<<1, 1024, 0, stream>>>(hs, tg, out);
}
